// SelfAttention_34643206209685
// MI455X (gfx1250) — matrix operand order repaired and checked by reading, not compiled, not run on hardware
//
#include <hip/hip_runtime.h>
#include <hip/hip_bf16.h>

typedef __attribute__((ext_vector_type(16))) __bf16 v16bf;
typedef __attribute__((ext_vector_type(8)))  __bf16 v8bf;
typedef __attribute__((ext_vector_type(8)))  float  v8f;

#define S_LEN  2048
#define DMODEL 768
#define NH     12
#define DH     64
#define BATCH  2

// ---- WMMA operand loaders (ISA 7.12.2 layouts, 16-bit elements) ----
// A-matrix 16x32 (MxK): lane L holds M=L%16; elems 0-7 = K[(L/16)*8 .. +7],
// elems 8-15 = K[16+(L/16)*8 .. +7].
__device__ __forceinline__ v16bf load_A(const __bf16* row, int hi) {
  v8bf c0 = *(const v8bf*)(row + hi * 8);
  v8bf c1 = *(const v8bf*)(row + 16 + hi * 8);
  v16bf r;
#pragma unroll
  for (int i = 0; i < 8; ++i) { r[i] = c0[i]; r[i + 8] = c1[i]; }
  return r;
}
// B-matrix 32x16 (KxN): lane L holds N=L%16; elems 0-7 = K[(L/16)*8 .. +7],
// elems 8-15 = K[16+(L/16)*8 .. +7] (k-major row of the B-source).
__device__ __forceinline__ v16bf load_B(const __bf16* row, int hi) {
  v8bf c0 = *(const v8bf*)(row + hi * 8);
  v8bf c1 = *(const v8bf*)(row + 16 + hi * 8);
  v16bf r;
#pragma unroll
  for (int i = 0; i < 8; ++i) { r[i] = c0[i]; r[i + 8] = c1[i]; }
  return r;
}

// ---- Kernel 0: fp32 -> bf16 conversion of X and packed [Wq;Wk;Wv] ----
__global__ void cvt_kernel(const float* __restrict__ X,
                           const float* __restrict__ Wq,
                           const float* __restrict__ Wk,
                           const float* __restrict__ Wv,
                           __bf16* __restrict__ xbf, __bf16* __restrict__ wbf) {
  const int NX = BATCH * S_LEN * DMODEL;
  const int NW = DMODEL * DMODEL;
  int idx = blockIdx.x * blockDim.x + threadIdx.x;
  if (idx < NX) {
    xbf[idx] = (__bf16)X[idx];
  } else if (idx < NX + 3 * NW) {
    int j = idx - NX;
    const float* src = (j < NW) ? Wq : (j < 2 * NW ? Wk : Wv);
    wbf[j] = (__bf16)src[j % NW];
  }
}

// ---- Kernel 1: fused QKV projection GEMM (4096 x 2304, k=768) ----
// Wave computes a 16(M) x 64(N) tile. q = X @ W^T + b: both A and B source
// rows are k-major.
__global__ __launch_bounds__(128)
void qkv_gemm(const __bf16* __restrict__ xbf, const __bf16* __restrict__ wbf,
              const float* __restrict__ bq, const float* __restrict__ bk,
              const float* __restrict__ bv,
              __bf16* __restrict__ q_ws, __bf16* __restrict__ k_ws,
              __bf16* __restrict__ vt_ws) {
  int lane = threadIdx.x & 31;
  int wave = threadIdx.x >> 5;
  int hi = lane >> 4;
  int ln = lane & 15;
  int nBase = blockIdx.x * 64;               // 0..2303 (one section/head per block)
  int mBase = blockIdx.y * 64 + wave * 16;   // 0..4095

  const __bf16* arow = xbf + (size_t)(mBase + ln) * DMODEL;
  v8f acc[4] = {};
  for (int k = 0; k < DMODEL; k += 32) {
    v16bf a = load_A(arow + k, hi);
#pragma unroll
    for (int j = 0; j < 4; ++j) {
      const __bf16* brow = wbf + (size_t)(nBase + j * 16 + ln) * DMODEL + k;
      v16bf b = load_B(brow, hi);
      acc[j] = __builtin_amdgcn_wmma_f32_16x16x32_bf16(
          false, a, false, b, (short)0, acc[j], false, false);
    }
  }

  int sec  = nBase / DMODEL;   // 0=Q, 1=K, 2=V (uniform per block)
  int nSec = nBase % DMODEL;
  const float* bias = (sec == 0) ? bq : (sec == 1 ? bk : bv);
#pragma unroll
  for (int j = 0; j < 4; ++j) {
    int col = nSec + j * 16 + ln;      // 0..767
    int h = col >> 6, d = col & 63;
    float bval = bias[col];
#pragma unroll
    for (int i = 0; i < 8; ++i) {
      int m = mBase + hi * 8 + i;      // global row (b*S + s)
      int b = m >> 11, s = m & (S_LEN - 1);
      __bf16 o = (__bf16)(acc[j][i] + bval);
      size_t bh = (size_t)(b * NH + h);
      if (sec == 0)      q_ws[(bh * S_LEN + s) * DH + d] = o;
      else if (sec == 1) k_ws[(bh * S_LEN + s) * DH + d] = o;
      else               vt_ws[(bh * DH + d) * S_LEN + s] = o;   // V transposed
    }
  }
}

// ---- Kernel 2: scores (+res, write out), softmax, P @ V ----
#define LPITCH 48   // bf16 elems per LDS row (96B, keeps b128 reads 16B aligned)
__global__ __launch_bounds__(256)
void attn_kernel(const __bf16* __restrict__ q_ws, const __bf16* __restrict__ k_ws,
                 const __bf16* __restrict__ vt_ws,
                 const float* __restrict__ res, const float* __restrict__ mask,
                 float* __restrict__ out_ctx, float* __restrict__ out_scores) {
  __shared__ __bf16 lds[8][16 * LPITCH];   // per-wave 16x32 P tile (D->A transpose)
  int lane = threadIdx.x & 31;
  int wave = threadIdx.x >> 5;
  int hi = lane >> 4, ln = lane & 15;
  int bh = blockIdx.y;            // 0..23
  int b  = bh / NH;
  int h  = bh % NH;
  int qBase = blockIdx.x * 128 + wave * 16;   // local q row (0..2047)

  const __bf16* qrow = q_ws + ((size_t)bh * S_LEN + qBase + ln) * DH;
  v16bf a0 = load_A(qrow, hi);
  v16bf a1 = load_A(qrow + 32, hi);

  const float scale = 0.125f;     // 1/sqrt(64)
  const float* resBase  = res       + ((size_t)bh * S_LEN + qBase) * S_LEN;
  float*       soutBase = out_scores + ((size_t)bh * S_LEN + qBase) * S_LEN;
  const float* maskB = mask + (size_t)b * S_LEN;

  float vmax[8];
#pragma unroll
  for (int i = 0; i < 8; ++i) vmax[i] = -3.0e38f;

  // ---- pass 1: scores = QK^T*scale + res ; store ; track rowmax(s+mask) ----
  for (int kt = 0; kt < S_LEN; kt += 16) {
    const __bf16* krow = k_ws + ((size_t)bh * S_LEN + kt + ln) * DH;
    v16bf b0 = load_B(krow, hi);
    v16bf b1 = load_B(krow + 32, hi);
    v8f c = {};
    c = __builtin_amdgcn_wmma_f32_16x16x32_bf16(false, a0, false, b0, (short)0, c, false, false);
    c = __builtin_amdgcn_wmma_f32_16x16x32_bf16(false, a1, false, b1, (short)0, c, false, false);
    float mk = maskB[kt + ln];
#pragma unroll
    for (int i = 0; i < 8; ++i) {
      size_t off = (size_t)(hi * 8 + i) * S_LEN + kt + ln;
      float s = c[i] * scale + resBase[off];
      soutBase[off] = s;                       // residual output (pre-mask)
      vmax[i] = fmaxf(vmax[i], s + mk);
    }
  }
  // cross-lane row max (C/D layout: 16-lane groups share the row set)
#pragma unroll
  for (int i = 0; i < 8; ++i) {
    float v = vmax[i];
    v = fmaxf(v, __shfl_xor(v, 1, 32));
    v = fmaxf(v, __shfl_xor(v, 2, 32));
    v = fmaxf(v, __shfl_xor(v, 4, 32));
    v = fmaxf(v, __shfl_xor(v, 8, 32));
    vmax[i] = v;
  }

  // ---- pass 2: p = exp(s+mask-max) ; ctx += P @ V ; rowsum ----
  float rsum[8] = {};
  v8f ctx[4] = {};
  __bf16* myLds = &lds[wave][0];

  for (int kb = 0; kb < S_LEN; kb += 32) {
#pragma unroll
    for (int sub = 0; sub < 2; ++sub) {
      int kt = kb + sub * 16;
      float mk = maskB[kt + ln];
#pragma unroll
      for (int i = 0; i < 8; ++i) {
        int row = hi * 8 + i;
        float s = soutBase[(size_t)row * S_LEN + kt + ln];   // L2-hot re-read
        float p = __expf(s + mk - vmax[i]);
        rsum[i] += p;
        myLds[row * LPITCH + sub * 16 + ln] = (__bf16)p;     // D-layout -> LDS
      }
    }
    asm volatile("s_wait_dscnt 0" ::: "memory");
    v16bf ap = load_A(myLds + ln * LPITCH, hi);              // A-layout reload
#pragma unroll
    for (int j = 0; j < 4; ++j) {
      const __bf16* vrow = vt_ws + ((size_t)bh * DH + j * 16 + ln) * S_LEN + kb;
      v16bf bv = load_B(vrow, hi);
      ctx[j] = __builtin_amdgcn_wmma_f32_16x16x32_bf16(
          false, ap, false, bv, (short)0, ctx[j], false, false);
    }
  }
  // cross-lane row sum, normalize
#pragma unroll
  for (int i = 0; i < 8; ++i) {
    float v = rsum[i];
    v += __shfl_xor(v, 1, 32);
    v += __shfl_xor(v, 2, 32);
    v += __shfl_xor(v, 4, 32);
    v += __shfl_xor(v, 8, 32);
    rsum[i] = 1.0f / v;
  }
#pragma unroll
  for (int j = 0; j < 4; ++j) {
#pragma unroll
    for (int i = 0; i < 8; ++i) {
      int q = qBase + hi * 8 + i;
      out_ctx[((size_t)(b * S_LEN) + q) * DMODEL + h * DH + j * 16 + ln] =
          ctx[j][i] * rsum[i];
    }
  }
}

extern "C" void kernel_launch(void* const* d_in, const int* in_sizes, int n_in,
                              void* d_out, int out_size, void* d_ws, size_t ws_size,
                              hipStream_t stream) {
  const float* X    = (const float*)d_in[0];   // hidden_states (2,2048,768)
  const float* mask = (const float*)d_in[1];   // (2,1,1,2048)
  const float* res  = (const float*)d_in[2];   // (2,12,2048,2048)
  const float* Wq   = (const float*)d_in[3];
  const float* bq   = (const float*)d_in[4];
  const float* Wk   = (const float*)d_in[5];
  const float* bk   = (const float*)d_in[6];
  const float* Wv   = (const float*)d_in[7];
  const float* bv   = (const float*)d_in[8];

  // workspace layout (bytes)
  char* ws = (char*)d_ws;
  __bf16* xbf   = (__bf16*)(ws);                 //  6,291,456 : X bf16
  __bf16* wbf   = (__bf16*)(ws +  6291456);      //  3,538,944 : [Wq;Wk;Wv] bf16
  __bf16* q_ws  = (__bf16*)(ws +  9830400);      //  6,291,456 : Q [B,H,S,Dh]
  __bf16* k_ws  = (__bf16*)(ws + 16121856);      //  6,291,456 : K [B,H,S,Dh]
  __bf16* vt_ws = (__bf16*)(ws + 22413312);      //  6,291,456 : V^T [B,H,Dh,S]

  float* out_ctx    = (float*)d_out;                               // (2,2048,768)
  float* out_scores = out_ctx + (size_t)BATCH * S_LEN * DMODEL;    // (2,12,2048,2048)

  {
    int total = BATCH * S_LEN * DMODEL + 3 * DMODEL * DMODEL;
    cvt_kernel<<<(total + 255) / 256, 256, 0, stream>>>(X, Wq, Wk, Wv, xbf, wbf);
  }
  {
    dim3 grid((3 * DMODEL) / 64, (BATCH * S_LEN) / 64);   // (36, 64)
    qkv_gemm<<<grid, 128, 0, stream>>>(xbf, wbf, bq, bk, bv, q_ws, k_ws, vt_ws);
  }
  {
    dim3 grid(S_LEN / 128, BATCH * NH);                   // (16, 24)
    attn_kernel<<<grid, 256, 0, stream>>>(q_ws, k_ws, vt_ws, res, mask,
                                          out_ctx, out_scores);
  }
}
